// DegradationAwareConv_86079734546992
// MI455X (gfx1250) — matrix operand order repaired and checked by reading, not compiled, not run on hardware
//
#include <hip/hip_runtime.h>

// ---- CDNA5 WMMA vector types ----
typedef __attribute__((ext_vector_type(16))) _Float16 v16h;
typedef __attribute__((ext_vector_type(8)))  _Float16 v8h;
typedef __attribute__((ext_vector_type(8)))  float    v8f;

#define KSZ   21
#define PAD   10
#define RROWS 64                  // output rows per workgroup region
#define RCOLS 64                  // output cols per workgroup region
#define AROWS (RROWS + 2*PAD)     // 84 staged input rows
#define ASTR  96                  // staged halves/row: covers tileX*16 + 47 (tileX<=3)
#define ASZ   (AROWS * ASTR)      // 8064 halves  (16128 B)
#define BSZ1  (KSZ * 16 * 32)     // 10752 halves per band (21504 B)
#define NTHREADS 128              // 4 waves; each wave owns a tileX column, 4 tileY tiles
// total static LDS: 16128 + 2*21504 = 59136 B  (< 64 KB)

union V16 { v16h v; v8h h[2]; };

__global__ __launch_bounds__(NTHREADS)
void dwconv21_wmma(const float* __restrict__ x,
                   const float* __restrict__ w,
                   float* __restrict__ out)
{
    __shared__ _Float16 sA[ASZ];        // f16 input tile with halo
    __shared__ _Float16 sB[2 * BSZ1];   // Toeplitz bands: [band][ky][n][u0..31]

    const int tid = threadIdx.x;
    const int blk = blockIdx.x;          // 96 * 64 blocks
    const int reg = blk & 63;            // 8x8 regions per channel-image
    const int bc  = blk >> 6;            // (b,c) flattened, 0..95
    const int b   = bc / 3;
    const int y0  = (reg >> 3) * RROWS;
    const int x0  = (reg & 7)  * RCOLS;

    // ---- stage input region (f32 -> f16), zero-padded halo ----
    const float* xim = x + (size_t)bc * (512 * 512);
    for (int i = tid; i < ASZ; i += NTHREADS) {
        int r = i / ASTR, c = i - r * ASTR;
        int gy = y0 - PAD + r, gx = x0 - PAD + c;
        float v = 0.f;
        if ((unsigned)gy < 512u && (unsigned)gx < 512u) v = xim[gy * 512 + gx];
        sA[i] = (_Float16)v;
    }

    // ---- build both Toeplitz bands ----
    // band 0: taps kx in [0,16):  B[u,j] = W[ky, u-j],    0 <= u-j <= 15
    // band 1: taps kx in [16,21): B[u,j] = W[ky, u+16-j], 16 <= u+16-j <= 20
    const float* wk = w + b * (KSZ * KSZ);
    for (int i = tid; i < 2 * BSZ1; i += NTHREADS) {
        int band = i / BSZ1;
        int r    = i - band * BSZ1;
        int ky   = r >> 9;
        int n    = (r >> 5) & 15;
        int u    = r & 31;
        int d    = u - n + (band ? 16 : 0);
        int lo   = band ? 16 : 0;
        int hi   = band ? (KSZ - 1) : 15;
        float v  = (d >= lo && d <= hi) ? wk[ky * KSZ + d] : 0.f;
        sB[i] = (_Float16)v;
    }
    __syncthreads();

    const int tx   = tid >> 5;           // wave id == tileX (0..3)
    const int lane = tid & 31;           // wave32
    const int lm   = lane & 15;
    const int lh   = lane >> 4;

    const v8f zacc = {0.f, 0.f, 0.f, 0.f, 0.f, 0.f, 0.f, 0.f};
    v8f acc[4] = {zacc, zacc, zacc, zacc};   // 4 tileY tiles per wave

    const int aColOff = (tx << 4) + (lh << 3);   // halves
    const int bOff    = (lm << 5) + (lh << 3);   // halves within a ky slab

    for (int ky = 0; ky < KSZ; ++ky) {
        // --- issue all LDS loads for this ky first, then the 8 WMMAs ---
        const _Float16* bb = sB + (ky << 9) + bOff;
        V16 Blo, Bhi;
        Blo.h[0] = *(const v8h*)(bb + 0);
        Blo.h[1] = *(const v8h*)(bb + 16);
        Bhi.h[0] = *(const v8h*)(bb + BSZ1 + 0);
        Bhi.h[1] = *(const v8h*)(bb + BSZ1 + 16);

        v8h a0[4], a1[4], a2[4];   // three 8-half chunks per tile (A windows overlap)
        #pragma unroll
        for (int t = 0; t < 4; ++t) {
            const _Float16* ab = sA + (t * 16 + ky + lm) * ASTR + aColOff;
            a0[t] = *(const v8h*)(ab + 0);    // u {0-7 | 8-15}
            a1[t] = *(const v8h*)(ab + 16);   // u {16-23 | 24-31}
            a2[t] = *(const v8h*)(ab + 32);   // u {32-39 | 40-47}
        }

        #pragma unroll
        for (int t = 0; t < 4; ++t) {
            V16 A1;
            A1.h[0] = a0[t]; A1.h[1] = a1[t];   // window u in [0,32)
            acc[t] = __builtin_amdgcn_wmma_f32_16x16x32_f16(
                false, A1.v, false, Blo.v, (short)0, acc[t], false, false);
        }
        #pragma unroll
        for (int t = 0; t < 4; ++t) {
            V16 A2;
            A2.h[0] = a1[t]; A2.h[1] = a2[t];   // window u in [16,48)
            acc[t] = __builtin_amdgcn_wmma_f32_16x16x32_f16(
                false, A2.v, false, Bhi.v, (short)0, acc[t], false, false);
        }
    }

    // ---- store: D layout is M = r + 8*lh, N = lm ----
    float* oim = out + (size_t)bc * (512 * 512);
    const int ox = x0 + (tx << 4) + lm;
    #pragma unroll
    for (int t = 0; t < 4; ++t) {
        int oy = y0 + t * 16 + (lh << 3);
        #pragma unroll
        for (int r = 0; r < 8; ++r)
            oim[(oy + r) * 512 + ox] = acc[t][r];
    }
}

extern "C" void kernel_launch(void* const* d_in, const int* in_sizes, int n_in,
                              void* d_out, int out_size, void* d_ws, size_t ws_size,
                              hipStream_t stream)
{
    (void)in_sizes; (void)n_in; (void)out_size; (void)d_ws; (void)ws_size;
    const float* x = (const float*)d_in[0];   // (32,3,512,512) f32
    const float* w = (const float*)d_in[1];   // (32,1,21,21)  f32
    float* out = (float*)d_out;               // (32,3,512,512) f32

    dim3 grid(96 * 64);      // 96 channel-images * 64 regions (64x64) each
    dim3 block(NTHREADS);    // 4 waves of 32
    dwconv21_wmma<<<grid, block, 0, stream>>>(x, w, out);
}
